// GAT_81887846465663
// MI455X (gfx1250) — compile-verified, matrix operand order checked by reading, not run on hardware
//
#include <hip/hip_runtime.h>
#include <hip/hip_bf16.h>

typedef __attribute__((ext_vector_type(16))) __bf16 v16bf;
typedef __attribute__((ext_vector_type(8)))  __bf16 v8bf;
typedef __attribute__((ext_vector_type(8)))  float  v8f;

// ---------------------------------------------------------------- utilities

__global__ void k_cvt_f32_bf16(const float* __restrict__ in, __bf16* __restrict__ out, int n) {
  int i = blockIdx.x * blockDim.x + threadIdx.x;
  if (i < n) out[i] = (__bf16)in[i];
}

__global__ void k_zero_f32(float* __restrict__ p, int n) {
  int i = blockIdx.x * blockDim.x + threadIdx.x;
  if (i < n) p[i] = 0.0f;
}

__global__ void k_init_softmax(float* __restrict__ emax, float* __restrict__ denom, int n) {
  int i = blockIdx.x * blockDim.x + threadIdx.x;
  if (i < n) { emax[i] = -__builtin_inff(); denom[i] = 0.0f; }
}

// ---------------------------------------------------- B-operand pre-packing
// Pack fp32 W[K,Nc] into the v_wmma_f32_16x16x32_bf16 B-operand layout so each
// lane's 16 bf16 values are contiguous (two global_load_b128 in the GEMM):
//   Bpk[ks][tn][lane][i], kk(lane,i) = (i<8) ? half*8+i : 16+half*8+(i-8),
//   k = ks*32+kk, n = tn*16 + (lane&15).
__global__ void k_pack_B(const float* __restrict__ W, __bf16* __restrict__ Bpk, int K, int Nc) {
  int t = blockIdx.x * blockDim.x + threadIdx.x;
  if (t >= K * Nc) return;
  const int ntc  = Nc >> 4;
  const int i    = t & 15;
  const int lane = (t >> 4) & 31;
  const int rest = t >> 9;
  const int tn   = rest % ntc;
  const int ks   = rest / ntc;
  const int half = lane >> 4, l16 = lane & 15;
  const int kk   = (i < 8) ? (half * 8 + i) : (16 + half * 8 + (i - 8));
  Bpk[t] = (__bf16)W[(size_t)(ks * 32 + kk) * Nc + tn * 16 + l16];
}

// ------------------------------------------------- bf16 WMMA GEMM, f32 accum
// C[row0.., Nc] = A * Bpk (+bias). One wave computes MT consecutive 16-row
// tiles of one 16-column tile: B registers reused MT times, fully unrolled,
// branch-free (no tail guard -> EXEC stays all-1s through every WMMA).
// Caller covers [row0, row0 + mtiles*16*MT) rows; any remainder is covered by
// a second launch of this template with MT=1. K = KSTEPS*32.
// A-operand: lane row m = lane&15 (+tile base), two contiguous 8-elem runs at
// k0 + half*8 and k0 + 16 + half*8. C/D: row = half*8 + r, col = lane&15.
template <int KSTEPS, int MT>
__global__ void k_wmma_gemm_bf16(const __bf16* __restrict__ A, const __bf16* __restrict__ Bpk,
                                 const float* __restrict__ bias, float* __restrict__ C,
                                 int row0, int mtiles, int Nc) {
  constexpr int K = KSTEPS * 32;
  const int lane = threadIdx.x & 31;
  const int wave = threadIdx.x >> 5;
  const int ntc  = Nc >> 4;
  const int tile  = blockIdx.x * (blockDim.x >> 5) + wave;
  const int total = mtiles * ntc;
  if (tile >= total) return;                 // wave-uniform
  const int tm = tile / ntc, tn = tile % ntc;
  const int mb = row0 + tm * (16 * MT), nbase = tn << 4;
  const int l16 = lane & 15, half = lane >> 4;

  v8f acc[MT];
#pragma unroll
  for (int mt = 0; mt < MT; ++mt) acc[mt] = (v8f){};

#pragma unroll
  for (int ks = 0; ks < KSTEPS; ++ks) {
    const int k0 = ks * 32;
    // B: 32 contiguous bytes per lane from the packed layout
    const v16bf b = *(const v16bf*)(Bpk + ((size_t)(ks * ntc + tn) * 32 + lane) * 16);
#pragma unroll
    for (int mt = 0; mt < MT; ++mt) {
      const int m = mb + mt * 16 + l16;
      const v8bf lo = *(const v8bf*)(A + (size_t)m * K + k0 + half * 8);
      const v8bf hi = *(const v8bf*)(A + (size_t)m * K + k0 + 16 + half * 8);
      v16bf a;
#pragma unroll
      for (int i = 0; i < 8; ++i) { a[i] = lo[i]; a[i + 8] = hi[i]; }
      acc[mt] = __builtin_amdgcn_wmma_f32_16x16x32_bf16(false, a, false, b,
                                                        (short)0, acc[mt], false, false);
    }
  }

  const float bb = bias ? bias[nbase + l16] : 0.0f;
#pragma unroll
  for (int mt = 0; mt < MT; ++mt) {
#pragma unroll
    for (int r = 0; r < 8; ++r) {
      const int row = mb + mt * 16 + half * 8 + r;
      C[(size_t)row * Nc + nbase + l16] = acc[mt][r] + bb;
    }
  }
}

// ---------------------------------------------- attention coefficients el/er
__global__ void k_attn_coeff(const float* __restrict__ feat, const float* __restrict__ al,
                             const float* __restrict__ ar, float* __restrict__ el,
                             float* __restrict__ er, int NH) {
  int t = blockIdx.x * blockDim.x + threadIdx.x;
  if (t >= NH) return;
  const int nidx = t >> 2, h = t & 3;
  const float* f  = feat + (size_t)nidx * 128 + h * 32;
  const float* a1 = al + h * 32;
  const float* a2 = ar + h * 32;
  float sl = 0.0f, sr = 0.0f;
#pragma unroll
  for (int i = 0; i < 32; ++i) { const float v = f[i]; sl += v * a1[i]; sr += v * a2[i]; }
  el[t] = sl; er[t] = sr;
}

// ------------------------------------------------------------ edge pipeline
__device__ __forceinline__ void atomicMaxF32(float* addr, float v) {
  // signed-int max for >=0, unsigned-int min for <0; valid with -inf init
  if (v >= 0.0f) atomicMax((int*)addr, __float_as_int(v));
  else           atomicMin((unsigned int*)addr, (unsigned int)__float_as_int(v));
}

__global__ void k_edge_logit_max(const float* __restrict__ el, const float* __restrict__ er,
                                 const int* __restrict__ src, const int* __restrict__ dst,
                                 float* __restrict__ evals, float* __restrict__ emax, int EH) {
  int t = blockIdx.x * blockDim.x + threadIdx.x;
  if (t >= EH) return;
  const int e = t >> 2, h = t & 3;
  float v = el[src[e] * 4 + h] + er[dst[e] * 4 + h];
  v = (v > 0.0f) ? v : 0.2f * v;             // leaky_relu, slope 0.2
  evals[t] = v;
  atomicMaxF32(&emax[dst[e] * 4 + h], v);
}

__global__ void k_fix_emax(float* __restrict__ emax, int n) {
  int i = blockIdx.x * blockDim.x + threadIdx.x;
  if (i < n) { const float m = emax[i]; if (!(m >= -3.0e38f)) emax[i] = 0.0f; }
}

__global__ void k_edge_exp_sum(float* __restrict__ evals, const float* __restrict__ emax,
                               const int* __restrict__ dst, float* __restrict__ denom, int EH) {
  int t = blockIdx.x * blockDim.x + threadIdx.x;
  if (t >= EH) return;
  const int e = t >> 2, h = t & 3;
  const float w = __expf(evals[t] - emax[dst[e] * 4 + h]);
  evals[t] = w;
  atomicAdd(&denom[dst[e] * 4 + h], w);
}

// one thread per (edge, channel); feat + acc are L2-resident (51 MB each)
__global__ void k_edge_scatter(const float* __restrict__ evals, const float* __restrict__ denom,
                               const float* __restrict__ feat, const int* __restrict__ src,
                               const int* __restrict__ dst, float* __restrict__ acc, long long EC) {
  long long t = (long long)blockIdx.x * blockDim.x + threadIdx.x;
  if (t >= EC) return;
  const int e = (int)(t >> 7);
  const int c = (int)(t & 127);
  const int h = c >> 5;
  const int d = dst[e];
  const float alpha = evals[e * 4 + h] / fmaxf(denom[d * 4 + h], 1e-9f);
  atomicAdd(&acc[(size_t)d * 128 + c], feat[(size_t)src[e] * 128 + c] * alpha);
}

__global__ void k_finalize_elu(float* __restrict__ h, const float* __restrict__ bias, int total) {
  int i = blockIdx.x * blockDim.x + threadIdx.x;
  if (i >= total) return;
  const float v = h[i] + bias[i & 127];
  h[i] = (v > 0.0f) ? v : (__expf(v) - 1.0f);  // elu(alpha=1)
}

// ------------------------------------------------------- predictor MLP 32->1
__global__ void k_predictor(const float* __restrict__ hp,
                            const int* __restrict__ ps, const int* __restrict__ pd,
                            const int* __restrict__ ns, const int* __restrict__ nd,
                            const float* __restrict__ W1, const float* __restrict__ b1,
                            const float* __restrict__ W2, const float* __restrict__ b2,
                            const float* __restrict__ W3, const float* __restrict__ b3,
                            float* __restrict__ out, int P) {
  __shared__ float sW1[1024], sW2[1024], sW3[32], sb1[32], sb2[32];
  for (int i = threadIdx.x; i < 1024; i += blockDim.x) { sW1[i] = W1[i]; sW2[i] = W2[i]; }
  if (threadIdx.x < 32) {
    sW3[threadIdx.x] = W3[threadIdx.x];
    sb1[threadIdx.x] = b1[threadIdx.x];
    sb2[threadIdx.x] = b2[threadIdx.x];
  }
  __syncthreads();

  const int t = blockIdx.x * blockDim.x + threadIdx.x;
  if (t >= 2 * P) return;
  const int p  = (t < P) ? t : t - P;
  const int si = (t < P) ? ps[p] : ns[p];
  const int di = (t < P) ? pd[p] : nd[p];
  const float* za = hp + (size_t)si * 32;
  const float* zb = hp + (size_t)di * 32;

  float z[32], y[32];
#pragma unroll
  for (int i = 0; i < 32; ++i) z[i] = za[i] * zb[i];
#pragma unroll
  for (int j = 0; j < 32; ++j) {
    float s = sb1[j];
#pragma unroll
    for (int i = 0; i < 32; ++i) s += z[i] * sW1[i * 32 + j];
    y[j] = fmaxf(s, 0.0f);
  }
#pragma unroll
  for (int j = 0; j < 32; ++j) {
    float s = sb2[j];
#pragma unroll
    for (int i = 0; i < 32; ++i) s += y[i] * sW2[i * 32 + j];
    z[j] = fmaxf(s, 0.0f);
  }
  float o = b3[0];
#pragma unroll
  for (int i = 0; i < 32; ++i) o += z[i] * sW3[i];
  out[t] = o;
}

// ------------------------------------------------------------------- launch

extern "C" void kernel_launch(void* const* d_in, const int* in_sizes, int n_in,
                              void* d_out, int out_size, void* d_ws, size_t ws_size,
                              hipStream_t stream) {
  const float* x       = (const float*)d_in[0];
  const int*   src     = (const int*)d_in[1];
  const int*   dst     = (const int*)d_in[2];
  const int*   pos_src = (const int*)d_in[3];
  const int*   pos_dst = (const int*)d_in[4];
  const int*   neg_src = (const int*)d_in[5];
  const int*   neg_dst = (const int*)d_in[6];
  const int N = in_sizes[0] / 128;
  const int E = in_sizes[1];
  const int P = in_sizes[3];

  const float* Wl[3]  = {(const float*)d_in[7],  (const float*)d_in[11], (const float*)d_in[15]};
  const float* all[3] = {(const float*)d_in[8],  (const float*)d_in[12], (const float*)d_in[16]};
  const float* arl[3] = {(const float*)d_in[9],  (const float*)d_in[13], (const float*)d_in[17]};
  const float* bl[3]  = {(const float*)d_in[10], (const float*)d_in[14], (const float*)d_in[18]};
  const float* Wp  = (const float*)d_in[19];
  const float* bp  = (const float*)d_in[20];
  const float* Wq1 = (const float*)d_in[21];
  const float* bq1 = (const float*)d_in[22];
  const float* Wq2 = (const float*)d_in[23];
  const float* bq2 = (const float*)d_in[24];
  const float* Wq3 = (const float*)d_in[25];
  const float* bq3 = (const float*)d_in[26];
  (void)n_in; (void)out_size; (void)ws_size;

  // -------- workspace carve (256B aligned); hcur doubles as scatter accumulator
  char* base = (char*)d_ws;
  auto carve = [&](size_t bytes) -> void* {
    void* p = (void*)base; base += (bytes + 255) & ~(size_t)255; return p;
  };
  float*  hcur  = (float*) carve((size_t)N * 128 * 4);
  __bf16* hb    = (__bf16*)carve((size_t)N * 128 * 2);
  float*  feat  = (float*) carve((size_t)N * 128 * 4);
  float*  el    = (float*) carve((size_t)N * 4 * 4);
  float*  er    = (float*) carve((size_t)N * 4 * 4);
  float*  emax  = (float*) carve((size_t)N * 4 * 4);
  float*  denom = (float*) carve((size_t)N * 4 * 4);
  float*  evals = (float*) carve((size_t)E * 4 * 4);
  __bf16* Bpk   = (__bf16*)carve((size_t)128 * 128 * 2);
  float*  hproj = (float*) carve((size_t)N * 32 * 4);

  const int T = 256;
  auto nb = [](long long n, int t) { return (unsigned)((n + t - 1) / t); };

  constexpr int MT = 5;
  const int mt16     = N / 16;                 // 16-row tiles (N % 16 == 0)
  const int mtiles   = mt16 / MT;              // branch-free super-tiles (1250)
  const int mrem     = mt16 % MT;              // remainder tiles (0 for N=100000)
  const int rem_row0 = mtiles * 16 * MT;

  // bulk + (optional) remainder GEMM launches, both branch-free inside
  auto gemm = [&](const __bf16* A, const float* bias, float* C, int Nc) {
    const int ntc = Nc >> 4;
    k_wmma_gemm_bf16<4, MT><<<nb((long long)mtiles * ntc, 8), T, 0, stream>>>(
        A, Bpk, bias, C, 0, mtiles, Nc);
    if (mrem > 0)
      k_wmma_gemm_bf16<4, 1><<<nb((long long)mrem * ntc, 8), T, 0, stream>>>(
          A, Bpk, bias, C, rem_row0, mrem, Nc);
  };

  for (int l = 0; l < 3; ++l) {
    const float* hin = (l == 0) ? x : hcur;
    k_cvt_f32_bf16<<<nb((long long)N * 128, T), T, 0, stream>>>(hin, hb, N * 128);
    k_pack_B<<<nb(128 * 128, T), T, 0, stream>>>(Wl[l], Bpk, 128, 128);

    gemm(hb, nullptr, feat, 128);

    k_attn_coeff<<<nb((long long)N * 4, T), T, 0, stream>>>(feat, all[l], arl[l], el, er, N * 4);
    k_zero_f32<<<nb((long long)N * 128, T), T, 0, stream>>>(hcur, N * 128);
    k_init_softmax<<<nb((long long)N * 4, T), T, 0, stream>>>(emax, denom, N * 4);

    k_edge_logit_max<<<nb((long long)E * 4, T), T, 0, stream>>>(el, er, src, dst, evals, emax, E * 4);
    k_fix_emax<<<nb((long long)N * 4, T), T, 0, stream>>>(emax, N * 4);
    k_edge_exp_sum<<<nb((long long)E * 4, T), T, 0, stream>>>(evals, emax, dst, denom, E * 4);
    k_edge_scatter<<<nb((long long)E * 128, T), T, 0, stream>>>(evals, denom, feat, src, dst,
                                                                hcur, (long long)E * 128);
    k_finalize_elu<<<nb((long long)N * 128, T), T, 0, stream>>>(hcur, bl[l], N * 128);
  }

  // projection 128 -> 32 with fused bias
  k_cvt_f32_bf16<<<nb((long long)N * 128, T), T, 0, stream>>>(hcur, hb, N * 128);
  k_pack_B<<<nb(128 * 32, T), T, 0, stream>>>(Wp, Bpk, 128, 32);
  gemm(hb, bp, hproj, 32);

  // predictor on pos (first P outputs) and neg (next P outputs)
  k_predictor<<<nb((long long)2 * P, T), T, 0, stream>>>(hproj, pos_src, pos_dst, neg_src, neg_dst,
                                                         Wq1, bq1, Wq2, bq2, Wq3, bq3,
                                                         (float*)d_out, P);
}
